// MultiHeadLatentAttention_11493332484466
// MI455X (gfx1250) — matrix operand order repaired and checked by reading, not compiled, not run on hardware
//
#include <hip/hip_runtime.h>
#include <hip/hip_bf16.h>

// ---------------- problem constants ----------------
#define D_MODEL 4096
#define N_HEADS 128
#define LATENT  512
#define ROPE    64
#define HEAD_DIM 32
#define BATCH   2
#define SEQ     1024
#define ROWS    (BATCH * SEQ)   // 2048 flattened (b,s) rows

// ---------------- WMMA plumbing ----------------
typedef __attribute__((ext_vector_type(16))) __bf16 v16bf;
typedef __attribute__((ext_vector_type(8)))  float  v8f;
typedef __attribute__((ext_vector_type(4)))  unsigned v4u;
typedef __attribute__((ext_vector_type(8)))  int      v8i;
typedef __attribute__((ext_vector_type(4)))  int      v4i;

union Frag { v16bf v; unsigned u[8]; };

__device__ inline unsigned short f2b(float f) {
  return __builtin_bit_cast(unsigned short, (__bf16)f);
}

__device__ inline v8f wmma_bf16(Frag a, Frag b, v8f c) {
  // D = A(16x32 bf16) * B(32x16 bf16) + C(16x16 f32)
  return __builtin_amdgcn_wmma_f32_16x16x32_bf16(
      /*neg_a=*/false, a.v, /*neg_b=*/false, b.v,
      /*c_mod=*/(short)0, c, /*reuse_a=*/false, /*reuse_b=*/false);
}

// ---------------- f32 -> bf16 cast ----------------
__global__ void cast_f32_bf16_kernel(const float* __restrict__ in,
                                     unsigned short* __restrict__ out, long n) {
  long i = (long)blockIdx.x * blockDim.x + threadIdx.x;
  long stride = (long)gridDim.x * blockDim.x;
  for (; i < n; i += stride) out[i] = f2b(in[i]);
}

// ---------------- TDM: 2D bf16 tile (tile_k x tile_m rows, row stride K) ----
// D# per CDNA5 ISA: group0 = {ctrl, lds_addr, global_addr,
// type=2}; group1 = {data_size=2B, tensor dims, tile dims, stride0=K}.
__device__ inline void tdm_load_tile_bf16(const unsigned short* gsrc,
                                          unsigned lds_off,
                                          int K, int tile_k, int tile_m) {
  unsigned long long ga = (unsigned long long)(const void*)gsrc;
  v4u g0;
  g0[0] = 1u;                                   // count=1, user descriptor
  g0[1] = lds_off;                              // lds_addr (bytes)
  g0[2] = (unsigned)(ga & 0xffffffffull);       // global_addr[31:0]
  g0[3] = (unsigned)((ga >> 32) & 0x01ffffffull) | (2u << 30);  // [56:32]|type=2

  const unsigned td0 = (unsigned)K;             // tensor_dim0 (elements)
  const unsigned td1 = 1u << 30;                // tensor_dim1: large, no OOB
  v8i g1;
  g1[0] = (int)(1u << 16);                      // data_size=1 (2 bytes)
  g1[1] = (int)((td0 & 0xffffu) << 16);         // [63:48] tensor_dim0 lo16
  g1[2] = (int)((td0 >> 16) | ((td1 & 0xffffu) << 16));
  g1[3] = (int)((td1 >> 16) | ((unsigned)tile_k << 16));  // tile_dim0
  g1[4] = (int)((unsigned)tile_m);              // tile_dim1, tile_dim2=0
  g1[5] = (int)((unsigned)K);                   // tensor_dim0_stride[31:0]
  g1[6] = 0;                                    // stride0 hi / stride1 lo
  g1[7] = 0;

  v4i z4 = {0, 0, 0, 0};
  v8i z8 = {0, 0, 0, 0, 0, 0, 0, 0};
  // clang-23 / therock-10.0 form: 6 args (g0, g1, g2, g3, extra, cpol)
  __builtin_amdgcn_tensor_load_to_lds(g0, g1, z4, z4, z8, 0);
}

// ---------------- GEMM: C[M,N] = A[M,K](bf16) * B[K,N](f32 -> bf16) --------
// Workgroup tile 128x64, 8 waves: wave (wm 0..3, wn 0..1) owns a 32x32 block
// -> 2 A frags x 2 B frags -> 4 WMMAs per 32-K step.  A panel staged by the
// Tensor Data Mover (wave 0 issues, s_wait_tensorcnt drains); B panel staged
// through VGPRs with f32->bf16 conversion.
__global__ __launch_bounds__(256)
void gemm_bf16_kernel(const unsigned short* __restrict__ A,
                      const float* __restrict__ B,
                      void* __restrict__ C,
                      int M, int N, int K, int out_f32) {
  __shared__ unsigned short sA[128 * 32];
  __shared__ unsigned short sB[32 * 64];

  const int tid  = threadIdx.x;
  const int wv   = tid >> 5;
  const int lane = tid & 31;
  const int wm = wv & 3, wn = wv >> 2;
  const int mloc = lane & 15, hgrp = lane >> 4;
  const int m_blk = blockIdx.y * 128, n_blk = blockIdx.x * 64;

  const int bidx = tid * 8;
  const int br = bidx >> 6, bc = bidx & 63;    // B tile 32x64

  const unsigned* sA32 = (const unsigned*)sA;
  const unsigned* sB32 = (const unsigned*)sB;
  const unsigned lds_a = (unsigned)(unsigned long long)(uintptr_t)&sA[0];

  v8f acc00 = {0.f,0.f,0.f,0.f,0.f,0.f,0.f,0.f};
  v8f acc01 = acc00, acc10 = acc00, acc11 = acc00;

  for (int kb = 0; kb < K; kb += 32) {
    // stage B: 8 f32 -> 8 bf16 per thread
    const float* gB = B + (size_t)(kb + br) * N + n_blk + bc;
    float4 f0 = *(const float4*)gB;
    float4 f1 = *(const float4*)(gB + 4);
    union { unsigned short s[8]; uint4 q; } pk;
    pk.s[0] = f2b(f0.x); pk.s[1] = f2b(f0.y); pk.s[2] = f2b(f0.z); pk.s[3] = f2b(f0.w);
    pk.s[4] = f2b(f1.x); pk.s[5] = f2b(f1.y); pk.s[6] = f2b(f1.z); pk.s[7] = f2b(f1.w);
    *(uint4*)&sB[br * 64 + bc] = pk.q;

    // stage A: TDM 2D tile copy, 128 rows x 64B, row stride K elements
    if (wv == 0) {
      tdm_load_tile_bf16(A + (size_t)m_blk * K + kb, lds_a, K, 32, 128);
      __builtin_amdgcn_s_wait_tensorcnt(0);
    }
    __syncthreads();

    // A fragments: lane&15 = M row; packed K pairs per ISA 16-bit A layout
    Frag a0, a1, b0, b1;
    const int am0 = wm * 32 + mloc;
    const int am1 = am0 + 16;
#pragma unroll
    for (int j = 0; j < 4; ++j) {
      a0.u[j]     = sA32[am0 * 16 + hgrp * 4 + j];
      a0.u[4 + j] = sA32[am0 * 16 + 8 + hgrp * 4 + j];
      a1.u[j]     = sA32[am1 * 16 + hgrp * 4 + j];
      a1.u[4 + j] = sA32[am1 * 16 + 8 + hgrp * 4 + j];
    }
    // B fragments: lane&15 = N col, dword j = K pair in table order
    const int nb0 = wn * 32 + mloc;  // N column
#pragma unroll
    for (int j = 0; j < 8; ++j) {
      const int kk = (j >> 2) * 16 + hgrp * 8 + (j & 3) * 2;
      b0.u[j] = (unsigned)sB[kk * 64 + nb0] | ((unsigned)sB[(kk + 1) * 64 + nb0] << 16);
      b1.u[j] = (unsigned)sB[kk * 64 + nb0 + 16] | ((unsigned)sB[(kk + 1) * 64 + nb0 + 16] << 16);
    }
    acc00 = wmma_bf16(a0, b0, acc00);
    acc01 = wmma_bf16(a0, b1, acc01);
    acc10 = wmma_bf16(a1, b0, acc10);
    acc11 = wmma_bf16(a1, b1, acc11);
    __syncthreads();
  }

  // C/D layout: lane&15 = N col, vgpr v -> M = v + 8*(lane>>4)
  const int row0 = m_blk + wm * 32 + hgrp * 8;
  const int col  = n_blk + wn * 32 + mloc;
  if (out_f32) {
    float* Cf = (float*)C;
#pragma unroll
    for (int v = 0; v < 8; ++v) {
      Cf[(size_t)(row0 + v) * N + col]           = acc00[v];
      Cf[(size_t)(row0 + v) * N + col + 16]      = acc01[v];
      Cf[(size_t)(row0 + 16 + v) * N + col]      = acc10[v];
      Cf[(size_t)(row0 + 16 + v) * N + col + 16] = acc11[v];
    }
  } else {
    unsigned short* Cb = (unsigned short*)C;
#pragma unroll
    for (int v = 0; v < 8; ++v) {
      Cb[(size_t)(row0 + v) * N + col]           = f2b(acc00[v]);
      Cb[(size_t)(row0 + v) * N + col + 16]      = f2b(acc01[v]);
      Cb[(size_t)(row0 + 16 + v) * N + col]      = f2b(acc10[v]);
      Cb[(size_t)(row0 + 16 + v) * N + col + 16] = f2b(acc11[v]);
    }
  }
}

// ---------------- flash attention ----------------
// grid (SEQ/128, N_HEADS, BATCH), 256 threads = 8 waves; each wave: 16 q rows.
// scores(16x16) = Qc(16x32)*KcT + Qr(16x64)*KrT  (3 chained WMMAs per tile),
// online softmax, P->bf16 via per-wave LDS reshape, out += P(16x32)*V(32x32).
__global__ __launch_bounds__(256)
void mla_attn_kernel(const unsigned short* __restrict__ qc,
                     const unsigned short* __restrict__ qr,
                     const unsigned short* __restrict__ kc,
                     const unsigned short* __restrict__ vc,
                     const unsigned short* __restrict__ kr,
                     unsigned short* __restrict__ ao) {
  __shared__ unsigned short sP[8][16 * 32];

  const int tid  = threadIdx.x;
  const int wv   = tid >> 5;
  const int lane = tid & 31;
  const int mloc = lane & 15, hgrp = lane >> 4;
  const int h  = blockIdx.y;
  const int bb = blockIdx.z;
  const int qbase = blockIdx.x * 128 + wv * 16;

  const unsigned* qc32 = (const unsigned*)qc;
  const unsigned* qr32 = (const unsigned*)qr;
  const unsigned* vc32 = (const unsigned*)vc;
  unsigned* sPw = (unsigned*)sP[wv];

  // Q fragments (A layout), loaded once
  Frag aq, ar0, ar1;
  {
    const size_t rq  = (size_t)(bb * SEQ + qbase + mloc) * N_HEADS + h;
    const size_t dwq = rq * 16;  // qc row, dwords (HEAD_DIM/2)
    const size_t dwr = rq * 32;  // qr row, dwords (ROPE/2)
#pragma unroll
    for (int j = 0; j < 4; ++j) {
      aq.u[j]      = qc32[dwq + hgrp * 4 + j];
      aq.u[4 + j]  = qc32[dwq + 8 + hgrp * 4 + j];
      ar0.u[j]     = qr32[dwr + hgrp * 4 + j];
      ar0.u[4 + j] = qr32[dwr + 8 + hgrp * 4 + j];
      ar1.u[j]     = qr32[dwr + 16 + hgrp * 4 + j];
      ar1.u[4 + j] = qr32[dwr + 24 + hgrp * 4 + j];
    }
  }

  float m_i[8], l_i[8];
  v8f o0 = {0.f,0.f,0.f,0.f,0.f,0.f,0.f,0.f};
  v8f o1 = o0;
#pragma unroll
  for (int v = 0; v < 8; ++v) { m_i[v] = -1e30f; l_i[v] = 0.f; }

  const float rscale = 0.10206207f;  // 1/sqrt(HEAD_DIM+ROPE)

  for (int kt = 0; kt < SEQ / 32; ++kt) {
    const int n0 = kt * 32;
    v8f sacc[2];
    sacc[0] = o0 - o0;  // zero
    sacc[1] = o0 - o0;

#pragma unroll
    for (int t = 0; t < 2; ++t) {
      const int nb = n0 + t * 16;
      Frag bkc, bkr0, bkr1;
#pragma unroll
      for (int j = 0; j < 8; ++j) {
        const int dk = (j >> 2) * 8 + hgrp * 4 + (j & 3);
        const size_t rk = (size_t)(bb * SEQ + nb + mloc) * N_HEADS + h;
        bkc.u[j] = ((const unsigned*)kc)[rk * 16 + dk];
        const size_t kd = (size_t)(bb * SEQ + nb + mloc) * (ROPE / 2);
        bkr0.u[j] = ((const unsigned*)kr)[kd + dk];
        bkr1.u[j] = ((const unsigned*)kr)[kd + 16 + dk];
      }
      sacc[t] = wmma_bf16(aq,  bkc,  sacc[t]);
      sacc[t] = wmma_bf16(ar0, bkr0, sacc[t]);
      sacc[t] = wmma_bf16(ar1, bkr1, sacc[t]);
    }

    // online softmax; row m = v + 8*hgrp, column chunk held by 16-lane group
    float alpha[8];
#pragma unroll
    for (int v = 0; v < 8; ++v) {
      sacc[0][v] *= rscale;
      sacc[1][v] *= rscale;
      float c = fmaxf(sacc[0][v], sacc[1][v]);
#pragma unroll
      for (int off = 1; off < 16; off <<= 1) c = fmaxf(c, __shfl_xor(c, off));
      const float mnew = fmaxf(m_i[v], c);
      alpha[v] = __expf(m_i[v] - mnew);
      m_i[v] = mnew;
    }
#pragma unroll
    for (int v = 0; v < 8; ++v) {
      const float p0 = __expf(sacc[0][v] - m_i[v]);
      const float p1 = __expf(sacc[1][v] - m_i[v]);
      float rs = p0 + p1;
#pragma unroll
      for (int off = 1; off < 16; off <<= 1) rs += __shfl_xor(rs, off);
      l_i[v] = l_i[v] * alpha[v] + rs;
      o0[v] *= alpha[v];
      o1[v] *= alpha[v];
      const int r = v + hgrp * 8;                // C-layout row
      sP[wv][r * 32 + mloc]      = f2b(p0);      // local key col t=0
      sP[wv][r * 32 + 16 + mloc] = f2b(p1);      // local key col t=1
    }

    // P fragment (A layout) back out of per-wave LDS tile (same-wave, in-order)
    Frag pf;
#pragma unroll
    for (int j = 0; j < 4; ++j) {
      pf.u[j]     = sPw[mloc * 16 + hgrp * 4 + j];
      pf.u[4 + j] = sPw[mloc * 16 + 8 + hgrp * 4 + j];
    }
    // V fragments: lane&15 = head-dim col, dword j = key pair in table order
    Frag bv0, bv1;
#pragma unroll
    for (int j = 0; j < 8; ++j) {
      const int kk = (j >> 2) * 16 + hgrp * 8 + (j & 3) * 2;
      const size_t vlo = ((size_t)(bb * SEQ + n0 + kk) * N_HEADS + h) * 32 + mloc;
      const size_t vhi = vlo + (size_t)N_HEADS * 32;
      bv0.u[j] = (unsigned)vc[vlo] | ((unsigned)vc[vhi] << 16);
      bv1.u[j] = (unsigned)vc[vlo + 16] | ((unsigned)vc[vhi + 16] << 16);
    }
    o0 = wmma_bf16(pf, bv0, o0);
    o1 = wmma_bf16(pf, bv1, o1);
  }

  // normalize + store (bf16) into ao[(b,s), h*32 + d]
#pragma unroll
  for (int v = 0; v < 8; ++v) {
    const float inv = 1.0f / l_i[v];
    const int srow = qbase + v + hgrp * 8;
    const size_t obase = ((size_t)(bb * SEQ + srow) * N_HEADS + h) * 32;
    ao[obase + mloc]      = f2b(o0[v] * inv);
    ao[obase + 16 + mloc] = f2b(o1[v] * inv);
  }
}

// ---------------- orchestration ----------------
extern "C" void kernel_launch(void* const* d_in, const int* in_sizes, int n_in,
                              void* d_out, int out_size, void* d_ws, size_t ws_size,
                              hipStream_t stream) {
  (void)in_sizes; (void)n_in; (void)out_size; (void)ws_size;

  const float* x       = (const float*)d_in[0];
  const float* Wq_down = (const float*)d_in[1];
  const float* Wq_up   = (const float*)d_in[2];
  const float* Wq_rope = (const float*)d_in[3];
  const float* Wkv_down= (const float*)d_in[4];
  const float* Wk_up   = (const float*)d_in[5];
  const float* Wv_up   = (const float*)d_in[6];
  const float* Wk_rope = (const float*)d_in[7];
  const float* Wo      = (const float*)d_in[8];
  float* out = (float*)d_out;

  // workspace layout (bf16 activations only; weights converted in-kernel)
  unsigned short* xb  = (unsigned short*)d_ws;              // 2048*4096
  unsigned short* cq  = xb  + (size_t)ROWS * D_MODEL;       // 2048*512
  unsigned short* ckv = cq  + (size_t)ROWS * LATENT;        // 2048*512
  unsigned short* qcb = ckv + (size_t)ROWS * LATENT;        // 2048*4096
  unsigned short* qrb = qcb + (size_t)ROWS * D_MODEL;       // 2048*8192
  unsigned short* kcb = qrb + (size_t)ROWS * N_HEADS * ROPE;// 2048*4096
  unsigned short* vcb = kcb + (size_t)ROWS * D_MODEL;       // 2048*4096
  unsigned short* krb = vcb + (size_t)ROWS * D_MODEL;       // 2048*64
  unsigned short* aob = krb + (size_t)ROWS * ROPE;          // 2048*4096

  // 1) x -> bf16
  cast_f32_bf16_kernel<<<4096, 256, 0, stream>>>(x, xb, (long)ROWS * D_MODEL);

  // 2) GEMMs: grid (N/64, M/128), M = 2048 rows
  dim3 blk(256);
  gemm_bf16_kernel<<<dim3(LATENT / 64, ROWS / 128), blk, 0, stream>>>(
      xb, Wq_down, cq, ROWS, LATENT, D_MODEL, 0);
  gemm_bf16_kernel<<<dim3(LATENT / 64, ROWS / 128), blk, 0, stream>>>(
      xb, Wkv_down, ckv, ROWS, LATENT, D_MODEL, 0);
  gemm_bf16_kernel<<<dim3(D_MODEL / 64, ROWS / 128), blk, 0, stream>>>(
      cq, Wq_up, qcb, ROWS, D_MODEL, LATENT, 0);
  gemm_bf16_kernel<<<dim3((N_HEADS * ROPE) / 64, ROWS / 128), blk, 0, stream>>>(
      xb, Wq_rope, qrb, ROWS, N_HEADS * ROPE, D_MODEL, 0);
  gemm_bf16_kernel<<<dim3(D_MODEL / 64, ROWS / 128), blk, 0, stream>>>(
      ckv, Wk_up, kcb, ROWS, D_MODEL, LATENT, 0);
  gemm_bf16_kernel<<<dim3(D_MODEL / 64, ROWS / 128), blk, 0, stream>>>(
      ckv, Wv_up, vcb, ROWS, D_MODEL, LATENT, 0);
  gemm_bf16_kernel<<<dim3(ROPE / 64, ROWS / 128), blk, 0, stream>>>(
      xb, Wk_rope, krb, ROWS, ROPE, D_MODEL, 0);

  // 3) attention
  mla_attn_kernel<<<dim3(SEQ / 128, N_HEADS, BATCH), blk, 0, stream>>>(
      qcb, qrb, kcb, vcb, krb, aob);

  // 4) output projection -> f32 d_out
  gemm_bf16_kernel<<<dim3(D_MODEL / 64, ROWS / 128), blk, 0, stream>>>(
      aob, Wo, out, ROWS, D_MODEL, D_MODEL, 1);
}
